// Network_790273982825
// MI455X (gfx1250) — matrix operand order repaired and checked by reading, not compiled, not run on hardware
//
#include <hip/hip_runtime.h>
#include <hip/hip_bf16.h>

typedef __attribute__((ext_vector_type(16))) _Float16 v16h;
typedef __attribute__((ext_vector_type(8)))  float    v8f;

#define BM 64
#define BN 128
#define BK 32

__device__ __forceinline__ int iclampi(int v, int lo, int hi) {
    return v < lo ? lo : (v > hi ? hi : v);
}

// ---------------------------------------------------------------------------
// Implicit-GEMM 3x3 conv (stride 1, zero pad 1) via WMMA f32 = f16 x f16.
// GEMM view: M = Cout, N = H*W, K = Cin*9.  Weights pre-converted to f16.
// Block: 256 threads = 8 waves. Block tile BM=64 x BN=128, K-step 32.
// Each wave owns a 32x32 macro-tile: 2 A-frags x 2 B-frags -> 4 WMMAs/K-step.
// LDS double-buffered (one barrier per K-step; staging overlaps WMMAs).
// A-stage: async global->LDS b128 DMA on interior tiles, batched VMEM else.
// B-stage: branch-free clamped im2col gather, batched loads then stores.
// All helpers are __forceinline__ so no outlining / scratch spills occur.
// ---------------------------------------------------------------------------
struct ConvCtx {
    const float*    in;
    const _Float16* wgt;
    int K, Cin, Cout, H, W, HW, coBase;
    int kStage, rowStage, p0, y0, x0;
    bool fastA;
    int  mAsync, kAsync;
};

template <int BUF>
__device__ __forceinline__ void stage_tile(const ConvCtx& c,
    _Float16 (&As)[2][BM][BK], _Float16 (&Bs)[2][BN][BK], int kc)
{
    const int kk  = kc + c.kStage;
    const bool kok = kk < c.K;

    // ---- A-stage ----
    if (c.fastA) {
        const _Float16* gsrc =
            c.wgt + (size_t)(c.coBase + c.mAsync) * c.K + kc + c.kAsync;
        const uint32_t ldsDst = (uint32_t)(uintptr_t)&As[BUF][c.mAsync][c.kAsync];
        asm volatile("global_load_async_to_lds_b128 %0, %1, off"
                     :: "v"(ldsDst), "v"(gsrc) : "memory");
    } else {
        _Float16 tw[BM / 8];
        unsigned wm = 0;
        int m = c.rowStage;
        #pragma unroll
        for (int it = 0; it < BM / 8; ++it, m += 8) {
            int co  = c.coBase + m;
            int coc = iclampi(co, 0, c.Cout - 1);
            int kkc = kok ? kk : c.K - 1;
            tw[it]  = c.wgt[(size_t)coc * c.K + kkc];
            wm     |= ((co < c.Cout && kok) ? 1u : 0u) << it;
        }
        m = c.rowStage;
        #pragma unroll
        for (int it = 0; it < BM / 8; ++it, m += 8)
            As[BUF][m][c.kStage] = ((wm >> it) & 1u) ? tw[it] : (_Float16)0.0f;
    }

    // ---- B-stage (im2col, zero pad): batched gather then store ----
    {
        int ci = kk / 9;
        int r  = kk - ci * 9;
        int ky = r / 3;
        int kx = r - ky * 3;
        const float* chan = c.in + (size_t)iclampi(ci, 0, c.Cin - 1) * c.HW;
        float    tv[BN / 8];
        unsigned okm = 0;
        int p = c.p0, y = c.y0, x = c.x0;
        #pragma unroll
        for (int it = 0; it < BN / 8; ++it) {
            int yy = y + ky - 1;
            int xx = x + kx - 1;
            bool ok = kok && (p < c.HW) &&
                      (yy >= 0) && (yy < c.H) && (xx >= 0) && (xx < c.W);
            int yyc = iclampi(yy, 0, c.H - 1);
            int xxc = iclampi(xx, 0, c.W - 1);
            tv[it]  = chan[yyc * c.W + xxc];
            okm    |= (ok ? 1u : 0u) << it;
            p += 8; x += 8;
            int wrap = (x >= c.W) ? 1 : 0;         // 8 < W always: single wrap
            x -= wrap ? c.W : 0;
            y += wrap;
        }
        #pragma unroll
        for (int it = 0; it < BN / 8; ++it) {
            Bs[BUF][c.rowStage + 8 * it][c.kStage] =
                ((okm >> it) & 1u) ? (_Float16)tv[it] : (_Float16)0.0f;
        }
    }
}

template <int BUF>
__device__ __forceinline__ void compute_tile(
    const _Float16 (&As)[2][BM][BK], const _Float16 (&Bs)[2][BN][BK],
    int waveM, int waveN, int mloc, int kbA, int kbB, v8f (&acc)[2][2])
{
    const _Float16* arow0 = &As[BUF][waveM * 32 +      mloc][0];
    const _Float16* arow1 = &As[BUF][waveM * 32 + 16 + mloc][0];
    const _Float16* brow0 = &Bs[BUF][waveN * 32 +      mloc][0];
    const _Float16* brow1 = &Bs[BUF][waveN * 32 + 16 + mloc][0];
    v16h a0, a1, b0, b1;
    #pragma unroll
    for (int i = 0; i < 8; ++i) {
        a0[i]     = arow0[kbA + i];
        a0[8 + i] = arow0[kbA + 8 + i];
        a1[i]     = arow1[kbA + i];
        a1[8 + i] = arow1[kbA + 8 + i];
    }
    #pragma unroll
    for (int i = 0; i < 16; ++i) {
        b0[i] = brow0[kbB + i];
        b1[i] = brow1[kbB + i];
    }
    acc[0][0] = __builtin_amdgcn_wmma_f32_16x16x32_f16(false, a0, false, b0,
                    (short)0, acc[0][0], false, false);
    acc[0][1] = __builtin_amdgcn_wmma_f32_16x16x32_f16(false, a0, false, b1,
                    (short)0, acc[0][1], false, false);
    acc[1][0] = __builtin_amdgcn_wmma_f32_16x16x32_f16(false, a1, false, b0,
                    (short)0, acc[1][0], false, false);
    acc[1][1] = __builtin_amdgcn_wmma_f32_16x16x32_f16(false, a1, false, b1,
                    (short)0, acc[1][1], false, false);
}

__global__ __launch_bounds__(256) void conv3x3_wmma(
    const float*    __restrict__ in,    // [Cin, H, W] f32
    const _Float16* __restrict__ wgt,   // [Cout, K] f16, K = Cin*9
    const float*    __restrict__ bias,  // [Cout] f32
    float*          __restrict__ out,   // [Cout, H, W] f32
    int Cin, int Cout, int H, int W, int relu)
{
    const int HW = H * W;
    const int K  = Cin * 9;
    const int tid  = threadIdx.x;
    const int lane = tid & 31;
    const int wave = tid >> 5;
    const int coBase = blockIdx.y * BM;
    const int pBase  = blockIdx.x * BN;
    const int waveM  = wave >> 2;   // 0..1  (32 rows each)
    const int waveN  = wave & 3;    // 0..3  (32 cols each)

    __shared__ _Float16 As[2][BM][BK];
    __shared__ _Float16 Bs[2][BN][BK];

    ConvCtx c;
    c.in = in; c.wgt = wgt;
    c.K = K; c.Cin = Cin; c.Cout = Cout; c.H = H; c.W = W; c.HW = HW;
    c.coBase = coBase;
    c.kStage   = tid & 31;
    c.rowStage = tid >> 5;
    c.p0 = pBase + c.rowStage;
    c.y0 = c.p0 / W;
    c.x0 = c.p0 - c.y0 * W;
    c.fastA  = ((K & (BK - 1)) == 0) && (coBase + BM <= Cout);
    c.mAsync = tid >> 2;
    c.kAsync = (tid & 3) * 8;

    const int mloc = lane & 15;
    const int kbA  = (lane < 16) ? 0 : 16;  // A: lanes 0-15 K 0..15; 16-31 K 16..31
    const int kbB  = (lane < 16) ? 0 : 16;  // B: lanes 0-15 K 0..15; 16-31 K 16..31

    v8f acc[2][2] = {};

    // ---- prologue ----
    stage_tile<0>(c, As, Bs, 0);
    if (c.fastA) asm volatile("s_wait_asynccnt 0x0" ::: "memory");
    __syncthreads();

    // ---- double-buffered main loop (x2 unroll: buffer index is literal) ----
    for (int k0 = 0; k0 < K; k0 += 2 * BK) {
        {   // step A: compute buf0, stage buf1 for k0+BK
            compute_tile<0>(As, Bs, waveM, waveN, mloc, kbA, kbB, acc);
            const int kn = k0 + BK;
            if (kn < K) {
                if (kn + BK < K) {      // prefetch two K-steps ahead
                    int cpf = iclampi(coBase + c.rowStage, 0, Cout - 1);
                    __builtin_prefetch(&wgt[(size_t)cpf * K + kn + BK + c.kStage], 0, 3);
                }
                stage_tile<1>(c, As, Bs, kn);
                if (c.fastA) asm volatile("s_wait_asynccnt 0x0" ::: "memory");
            }
            __syncthreads();
        }
        if (k0 + BK < K) {  // step B: compute buf1, stage buf0 for k0+2*BK
            compute_tile<1>(As, Bs, waveM, waveN, mloc, kbA, kbB, acc);
            const int kn = k0 + 2 * BK;
            if (kn < K) {
                if (kn + BK < K) {
                    int cpf = iclampi(coBase + c.rowStage, 0, Cout - 1);
                    __builtin_prefetch(&wgt[(size_t)cpf * K + kn + BK + c.kStage], 0, 3);
                }
                stage_tile<0>(c, As, Bs, kn);
                if (c.fastA) asm volatile("s_wait_asynccnt 0x0" ::: "memory");
            }
            __syncthreads();
        }
    }

    // ---- writeback: lane l, reg r -> M = r + (l<16?0:8), N = l&15 ----
    const int nloc = lane & 15;
    #pragma unroll
    for (int sn = 0; sn < 2; ++sn) {
        int p = pBase + waveN * 32 + sn * 16 + nloc;
        if (p >= HW) continue;
        #pragma unroll
        for (int sm = 0; sm < 2; ++sm) {
            #pragma unroll
            for (int r = 0; r < 8; ++r) {
                int m  = waveM * 32 + sm * 16 + ((lane < 16) ? r : r + 8);
                int co = coBase + m;
                if (co < Cout) {
                    float v = acc[sm][sn][r] + bias[co];
                    if (relu) v = fmaxf(v, 0.0f);
                    out[(size_t)co * HW + p] = v;
                }
            }
        }
    }
}

// ---------------------------------------------------------------------------
// f32 -> f16 weight pre-conversion
// ---------------------------------------------------------------------------
__global__ void f32_to_f16_kernel(const float* __restrict__ in,
                                  _Float16* __restrict__ out, int n)
{
    int i = blockIdx.x * blockDim.x + threadIdx.x;
    if (i < n) out[i] = (_Float16)in[i];
}

// ---------------------------------------------------------------------------
// 2x2 average pool
// ---------------------------------------------------------------------------
__global__ void avgpool2_kernel(const float* __restrict__ in, float* __restrict__ out,
                                int C, int H, int W)
{
    int Ho = H >> 1, Wo = W >> 1;
    int total = C * Ho * Wo;
    int idx = blockIdx.x * blockDim.x + threadIdx.x;
    if (idx >= total) return;
    int x = idx % Wo;
    int t = idx / Wo;
    int y = t % Ho;
    int c = t / Ho;
    const float* b = in + (size_t)c * H * W + (2 * y) * W + 2 * x;
    out[idx] = 0.25f * (b[0] + b[1] + b[W] + b[W + 1]);
}

// ---------------------------------------------------------------------------
// bilinear x2, align_corners=True
// ---------------------------------------------------------------------------
__global__ void up2_kernel(const float* __restrict__ in, float* __restrict__ out,
                           int C, int H, int W)
{
    int Ho = 2 * H, Wo = 2 * W;
    int total = C * Ho * Wo;
    int idx = blockIdx.x * blockDim.x + threadIdx.x;
    if (idx >= total) return;
    int x = idx % Wo;
    int t = idx / Wo;
    int y = t % Ho;
    int c = t / Ho;
    float py = y * ((float)(H - 1) / (float)(Ho - 1));
    float px = x * ((float)(W - 1) / (float)(Wo - 1));
    int iy = (int)py; if (iy > H - 2) iy = H - 2;
    int ix = (int)px; if (ix > W - 2) ix = W - 2;
    float ty = py - iy, tx = px - ix;
    const float* base = in + (size_t)c * H * W;
    float a = base[iy * W + ix],        b = base[iy * W + ix + 1];
    float cc = base[(iy + 1) * W + ix], d = base[(iy + 1) * W + ix + 1];
    float top = a * (1.0f - tx) + b * tx;
    float bot = cc * (1.0f - tx) + d * tx;
    out[idx] = top * (1.0f - ty) + bot * ty;
}

// ---------------------------------------------------------------------------
// elementwise add (out may alias a)
// ---------------------------------------------------------------------------
__global__ void add_kernel(const float* __restrict__ a, const float* __restrict__ b,
                           float* __restrict__ out, int n)
{
    int idx = blockIdx.x * blockDim.x + threadIdx.x;
    if (idx < n) out[idx] = a[idx] + b[idx];
}

// ---------------------------------------------------------------------------
// Fused adaptive separable convolution for both frames, replicate padding
// folded into clamped indexing.
// ---------------------------------------------------------------------------
__device__ __forceinline__ void sep_accum(const float* __restrict__ T,
                                          const float* __restrict__ V,
                                          const float* __restrict__ Hm,
                                          int idx, int x, int y, int H, int W,
                                          float acc[3])
{
    const int HW = H * W;
    float hk[51];
    #pragma unroll
    for (int j = 0; j < 51; ++j) hk[j] = Hm[j * HW + idx];
    for (int i = 0; i < 51; ++i) {
        float vi = V[i * HW + idx];
        int yy = iclampi(y + i - 25, 0, H - 1);
        #pragma unroll
        for (int c = 0; c < 3; ++c) {
            const float* row = T + c * HW + yy * W;
            float s = 0.0f;
            #pragma unroll
            for (int j = 0; j < 51; ++j) {
                int xx = iclampi(x + j - 25, 0, W - 1);
                s = fmaf(row[xx], hk[j], s);
            }
            acc[c] = fmaf(vi, s, acc[c]);
        }
    }
}

__global__ void sepconv_kernel(const float* __restrict__ t1, const float* __restrict__ t2,
                               const float* __restrict__ v1, const float* __restrict__ h1,
                               const float* __restrict__ v2, const float* __restrict__ h2,
                               float* __restrict__ out, int H, int W)
{
    int idx = blockIdx.x * blockDim.x + threadIdx.x;
    if (idx >= H * W) return;
    int x = idx % W, y = idx / W;
    float acc[3] = {0.0f, 0.0f, 0.0f};
    sep_accum(t1, v1, h1, idx, x, y, H, W, acc);
    sep_accum(t2, v2, h2, idx, x, y, H, W, acc);
    int HW = H * W;
    #pragma unroll
    for (int c = 0; c < 3; ++c) out[c * HW + idx] = acc[c];
}

// ---------------------------------------------------------------------------
// Host-side orchestration helpers
// ---------------------------------------------------------------------------
static inline void launch_conv(hipStream_t s, const float* in, const _Float16* w,
                               const float* b, float* out,
                               int Cin, int Cout, int H, int W, bool relu)
{
    dim3 grid((H * W + BN - 1) / BN, (Cout + BM - 1) / BM);
    conv3x3_wmma<<<grid, 256, 0, s>>>(in, w, b, out, Cin, Cout, H, W, relu ? 1 : 0);
}

static inline void launch_pool(hipStream_t s, const float* in, float* out, int C, int H, int W)
{
    int n = C * (H / 2) * (W / 2);
    avgpool2_kernel<<<(n + 255) / 256, 256, 0, s>>>(in, out, C, H, W);
}

static inline void launch_up(hipStream_t s, const float* in, float* out, int C, int H, int W)
{
    int n = C * 2 * H * 2 * W;
    up2_kernel<<<(n + 255) / 256, 256, 0, s>>>(in, out, C, H, W);
}

static inline void launch_add(hipStream_t s, const float* a, const float* b, float* out, int n)
{
    add_kernel<<<(n + 255) / 256, 256, 0, s>>>(a, b, out, n);
}

extern "C" void kernel_launch(void* const* d_in, const int* in_sizes, int n_in,
                              void* d_out, int out_size, void* d_ws, size_t ws_size,
                              hipStream_t stream)
{
    (void)n_in; (void)out_size; (void)ws_size;

    // Flattened param order (dict insertion order of setup_inputs):
    //  0: tenOne, 1: tenTwo, then (w,b) pairs:
    //  conv1:2..7  conv2:8..13  conv3:14..19  conv4:20..25  conv5:26..31
    //  deconv5:32..37 deconv4:38..43 deconv3:44..49 deconv2:50..55
    //  up5:56,57 up4:58,59 up3:60,61 up2:62,63
    //  v1:64..71 v2:72..79 h1:80..87 h2:88..95
    const float* P[96];
    for (int i = 0; i < 96; ++i) P[i] = (const float*)d_in[i];
    const float* tenOne = P[0];
    const float* tenTwo = P[1];
    float* out = (float*)d_out;

    const int H0 = 256, W0 = 448;
    const int HW0 = H0 * W0;                 // 114688
    const int H2 = 128, W2 = 224;
    const int H3 = 64,  W3 = 112;
    const int H4 = 32,  W4 = 56;
    const int H5 = 16,  W5 = 28;
    const int H6 = 8,   W6 = 14;

    // ---- workspace layout (float units), bump allocated, deterministic ----
    float* ws = (float*)d_ws;
    size_t off = 0;
    auto alloc = [&](size_t nflt) {
        float* p = ws + off;
        off += (nflt + 63) & ~((size_t)63);
        return p;
    };
    const size_t SBIG = (size_t)32 * HW0;    // 3,670,016 floats: largest map
    float* S0 = alloc(SBIG);
    float* S1 = alloc(SBIG);
    float* S2 = alloc(SBIG);
    float* c2b  = alloc((size_t)64 * H2 * W2);
    float* c3b  = alloc((size_t)128 * H3 * W3);
    float* c4b  = alloc((size_t)256 * H4 * W4);
    float* c5b  = alloc((size_t)512 * H5 * W5);
    float* comb = alloc((size_t)64 * H2 * W2);
    float* big  = alloc((size_t)51 * HW0);
    float* v1m  = alloc((size_t)51 * HW0);
    float* v2m  = alloc((size_t)51 * HW0);
    float* h1m  = alloc((size_t)51 * HW0);
    float* h2m  = alloc((size_t)51 * HW0);

    // ---- pre-convert all weight tensors (even indices >= 2) to f16 ----
    const _Float16* Wh[96] = {};
    for (int i = 2; i < 96; i += 2) {
        int n = in_sizes[i];
        _Float16* dst = (_Float16*)alloc(((size_t)n + 1) / 2);
        f32_to_f16_kernel<<<(n + 255) / 256, 256, 0, stream>>>(P[i], dst, n);
        Wh[i] = dst;
    }

    // basic(): three conv+relu. t1 != in; t2 != t1; out may alias t1.
    auto basic3 = [&](const float* in, int pi, float* t1, float* t2, float* outp,
                      int cin, int cout, int Hh, int Ww) {
        launch_conv(stream, in, Wh[pi + 0], P[pi + 1], t1, cin,  cout, Hh, Ww, true);
        launch_conv(stream, t1, Wh[pi + 2], P[pi + 3], t2, cout, cout, Hh, Ww, true);
        launch_conv(stream, t2, Wh[pi + 4], P[pi + 5], outp, cout, cout, Hh, Ww, true);
    };

    // ---- encoder ----
    hipMemcpyAsync(S0,           tenOne, (size_t)3 * HW0 * sizeof(float),
                   hipMemcpyDeviceToDevice, stream);
    hipMemcpyAsync(S0 + 3 * HW0, tenTwo, (size_t)3 * HW0 * sizeof(float),
                   hipMemcpyDeviceToDevice, stream);

    basic3(S0, 2,  S1, S2, S1,   6,  32, H0, W0);              // c1 = S1
    launch_pool(stream, S1, S0, 32, H0, W0);                   // p1 = S0
    basic3(S0, 8,  S1, S2, c2b, 32,  64, H2, W2);              // c2
    launch_pool(stream, c2b, S0, 64, H2, W2);                  // p2 = S0
    basic3(S0, 14, S1, S2, c3b, 64, 128, H3, W3);              // c3
    launch_pool(stream, c3b, S0, 128, H3, W3);                 // p3 = S0
    basic3(S0, 20, S1, S2, c4b, 128, 256, H4, W4);             // c4
    launch_pool(stream, c4b, S0, 256, H4, W4);                 // p4 = S0
    basic3(S0, 26, S1, S2, c5b, 256, 512, H5, W5);             // c5
    launch_pool(stream, c5b, S0, 512, H5, W5);                 // p5 = S0

    // ---- decoder ----
    basic3(S0, 32, S1, S2, S1, 512, 512, H6, W6);              // deconv5 -> S1
    launch_up(stream, S1, S0, 512, H6, W6);
    launch_conv(stream, S0, Wh[56], P[57], S1, 512, 512, H5, W5, true); // d5 = S1
    launch_add(stream, S1, c5b, S1, 512 * H5 * W5);            // s5 = S1

    basic3(S1, 38, S0, S2, S0, 512, 256, H5, W5);              // deconv4 -> S0
    launch_up(stream, S0, S1, 256, H5, W5);
    launch_conv(stream, S1, Wh[58], P[59], S0, 256, 256, H4, W4, true); // d4 = S0
    launch_add(stream, S0, c4b, S0, 256 * H4 * W4);            // s4 = S0

    basic3(S0, 44, S1, S2, S1, 256, 128, H4, W4);              // deconv3 -> S1
    launch_up(stream, S1, S0, 128, H4, W4);
    launch_conv(stream, S0, Wh[60], P[61], S1, 128, 128, H3, W3, true); // d3 = S1
    launch_add(stream, S1, c3b, S1, 128 * H3 * W3);            // s3 = S1

    basic3(S1, 50, S0, S2, S0, 128, 64, H3, W3);               // deconv2 -> S0
    launch_up(stream, S0, S1, 64, H3, W3);
    launch_conv(stream, S1, Wh[62], P[63], S0, 64, 64, H2, W2, true);   // d2 = S0
    launch_add(stream, S0, c2b, comb, 64 * H2 * W2);           // comb

    // ---- four kernel-prediction subnets ----
    const int subBase[4] = {64, 72, 80, 88};
    float* subOut[4]     = {v1m, v2m, h1m, h2m};
    for (int k = 0; k < 4; ++k) {
        int pi = subBase[k];
        launch_conv(stream, comb, Wh[pi + 0], P[pi + 1], S0, 64, 64, H2, W2, true);
        launch_conv(stream, S0,   Wh[pi + 2], P[pi + 3], S1, 64, 64, H2, W2, true);
        launch_conv(stream, S1,   Wh[pi + 4], P[pi + 5], S0, 64, 51, H2, W2, true);
        launch_up(stream, S0, big, 51, H2, W2);
        launch_conv(stream, big, Wh[pi + 6], P[pi + 7], subOut[k], 51, 51, H0, W0, false);
    }

    // ---- adaptive separable convolution (both frames fused) ----
    sepconv_kernel<<<(HW0 + 255) / 256, 256, 0, stream>>>(
        tenOne, tenTwo, v1m, h1m, v2m, h2m, out, H0, W0);
}
